// CenterHeadIoU_1d_34961033789446
// MI455X (gfx1250) — matrix operand order repaired and checked by reading, not compiled, not run on hardware
//
#include <hip/hip_runtime.h>
#include <hip/hip_bf16.h>
#include <stdint.h>

// ---------------- problem constants ----------------
#define B_SZ   32
#define C_IN   128
#define NPTS   8192
#define HC     64
#define TN     128            // positions per workgroup
#define CINP   136            // padded K stride (halves) for 128-K activations (row = 272B, 16B aligned)
#define HCP    72             // padded K stride (halves) for 64-K activations (row = 144B, 16B aligned)
#define EPSV   1e-5f

typedef _Float16 half_t;
typedef __attribute__((ext_vector_type(16))) _Float16 v16h;
typedef __attribute__((ext_vector_type(8)))  float    v8f;
typedef __attribute__((ext_vector_type(4)))  uint32_t v4u;

// ---------------- LDS layout (bytes, all 16B aligned) ----------------
#define OFF_XT    0            // [TN][CINP] f16              = 34816
#define OFF_YT    34816        // [TN][HCP]  f16              = 18432
#define OFF_ZT    53248        // [TN][HCP]  f16              = 18432
#define OFF_WSH   71680        // 16 frags x 1024B (swizzled) = 16384
#define OFF_WH1   88064        // 6 x 8 frags x 1024B         = 49152
#define OFF_WF    137216       // 6 x 2 frags x 1024B         = 12288
#define OFF_BY    149504       // [64]  f32
#define OFF_SY    149760       // [64]  f32
#define OFF_BZ    150016       // [384] f32
#define OFF_SZ    151552       // [384] f32
#define OFF_BF    153088       // [96]  f32
#define SMEM_BYTES 153472

__constant__ int d_COUT[6] = {3, 2, 1, 3, 2, 1};
__constant__ int d_HOFF[6] = {0, 3, 5, 6, 9, 11};

struct Params {
    const float* ct;
    const float* sh_w; const float* sh_b; const float* sh_g; const float* sh_beta;
    const float* sh_mean; const float* sh_var;
    const float* h_w1; const float* h_b1; const float* h_g1; const float* h_beta1;
    const float* h_mean1; const float* h_var1;
    const float* fw[6]; const float* fb[6];
    float* out;
};

// ---- A-fragment swizzled storage ----
// Each fragment = 1024B: lane owns 32B (2 x 16B granules), granule-XOR swizzled
// so per-instruction b128 reads hit every bank exactly twice (the minimum).
__device__ __forceinline__ int frag_byteoff(int frag, int lane, int e) {
    int x = 2 * lane + (e >> 3);          // logical 16B granule
    int g = x ^ ((x >> 4) & 3);           // bank swizzle
    return frag * 1024 + g * 16 + (e & 7) * 2;
}

// logical (row m-within-tile, k-within-32-chunk) for A 16x32 f16 fragment
// element e of lane: K = 16*hi + e (contiguous, matches B fragment order)
__device__ __forceinline__ void frag_elem(int lane, int e, int& m, int& kic) {
    int hi2 = lane >> 4;
    m = lane & 15;
    kic = hi2 * 16 + e;
}

__device__ __forceinline__ v16h load_afrag(const char* fragbase, int ga0, int ga1) {
    union { v16h v; v4u q[2]; } a;
    a.q[0] = *(const v4u*)(fragbase + ga0);
    a.q[1] = *(const v4u*)(fragbase + ga1);
    return a.v;
}

// B 32x16 f16 fragment: lane = hi*16 + n; element e holds K = hi*16 + e.
// Activations stored transposed [col][channel] -> 16 contiguous halves, 16B aligned.
__device__ __forceinline__ v16h load_bfrag(const half_t* xr) {
    union { v16h v; v4u q[2]; } bfr;
    bfr.q[0] = *(const v4u*)(xr);
    bfr.q[1] = *(const v4u*)(xr + 8);
    return bfr.v;
}

__device__ __forceinline__ v8f wmma_f16(v16h a, v16h b, v8f c) {
    return __builtin_amdgcn_wmma_f32_16x16x32_f16(
        false, a, false, b, (short)0, c, false, false);
}

__global__ __launch_bounds__(256)
void center_head_fused(Params p) {
    extern __shared__ char smem[];
    half_t* Xt = (half_t*)(smem + OFF_XT);   // [TN][CINP]
    half_t* Yt = (half_t*)(smem + OFF_YT);   // [TN][HCP]
    half_t* Zt = (half_t*)(smem + OFF_ZT);   // [TN][HCP]
    float* biasY  = (float*)(smem + OFF_BY);
    float* scaleY = (float*)(smem + OFF_SY);
    float* biasZ  = (float*)(smem + OFF_BZ);
    float* scaleZ = (float*)(smem + OFF_SZ);
    float* biasF  = (float*)(smem + OFF_BF);

    const int tid = threadIdx.x;
    const int b   = blockIdx.y;
    const int n0  = blockIdx.x * TN;

    // ---- phase A: fold BN into per-channel scale/bias ----
    if (tid < HC) {
        float a = p.sh_g[tid] * rsqrtf(p.sh_var[tid] + EPSV);
        scaleY[tid] = a;
        biasY[tid]  = (p.sh_b[tid] - p.sh_mean[tid]) * a + p.sh_beta[tid];
    }
    for (int i = tid; i < 6 * HC; i += 256) {
        float a = p.h_g1[i] * rsqrtf(p.h_var1[i] + EPSV);
        scaleZ[i] = a;
        biasZ[i]  = (p.h_b1[i] - p.h_mean1[i]) * a + p.h_beta1[i];
    }
#pragma unroll
    for (int h = 0; h < 6; ++h) {
        if (tid < 16)
            biasF[h * 16 + tid] = (tid < d_COUT[h]) ? p.fb[h][tid] : 0.0f;
    }
    __syncthreads();

    // ---- phase B: convert weights -> f16, BN scale folded, fragment-swizzled ----
    for (int i = tid; i < HC * C_IN; i += 256) {          // Wsh: 16 frags (4 mt x 4 kc)
        int frag = i >> 9;
        int lane = (i >> 4) & 31;
        int e    = i & 15;
        int m, kic; frag_elem(lane, e, m, kic);
        int o = (frag >> 2) * 16 + m;
        int k = (frag & 3) * 32 + kic;
        *(half_t*)(smem + OFF_WSH + frag_byteoff(frag, lane, e)) =
            (half_t)(p.sh_w[o * C_IN + k] * scaleY[o]);
    }
    for (int i = tid; i < 6 * HC * HC; i += 256) {        // Wh1: 6 x 8 frags (4 mt x 2 kc)
        int h    = i >> 12;
        int r    = i & 4095;
        int frag = r >> 9;
        int lane = (r >> 4) & 31;
        int e    = r & 15;
        int m, kic; frag_elem(lane, e, m, kic);
        int o = (frag >> 1) * 16 + m;
        int k = (frag & 1) * 32 + kic;
        *(half_t*)(smem + OFF_WH1 + h * 8192 + frag_byteoff(frag, lane, e)) =
            (half_t)(p.h_w1[(h * HC + o) * HC + k] * scaleZ[h * HC + o]);
    }
#pragma unroll
    for (int h = 0; h < 6; ++h) {                          // Wf: 6 x 2 frags, rows >= c_out zeroed
        for (int i = tid; i < 1024; i += 256) {
            int frag = i >> 9;
            int lane = (i >> 4) & 31;
            int e    = i & 15;
            int m, kic; frag_elem(lane, e, m, kic);
            int k = frag * 32 + kic;
            float v = (m < d_COUT[h]) ? p.fw[h][m * HC + k] : 0.0f;
            *(half_t*)(smem + OFF_WF + h * 2048 + frag_byteoff(frag, lane, e)) = (half_t)v;
        }
    }
    __syncthreads();

    // ---- per-wave slab: wave owns 16 columns; stages chain wave-locally ----
    const int wave  = tid >> 5;
    const int lid   = tid & 31;
    const int hi    = lid >> 4;
    const int ln    = lid & 15;
    const int nbase = wave * 16;
    const int ga0 = ((2 * lid)     ^ (((2 * lid)     >> 4) & 3)) << 4;  // hoisted A granule offsets
    const int ga1 = ((2 * lid + 1) ^ (((2 * lid + 1) >> 4) & 3)) << 4;

    // load this wave's X columns -> LDS transposed [col][channel], f32->f16, 2 channels/store
    {
        const int ncol = nbase + ln;
        const float* src = p.ct + ((size_t)b * C_IN + hi * 64) * NPTS + n0 + ncol;
        half_t* dst = Xt + ncol * CINP + hi * 64;
#pragma unroll 8
        for (int c = 0; c < 64; c += 2) {
            union { half_t h2[2]; uint32_t u; } pk;
            pk.h2[0] = (half_t)src[(size_t)c * NPTS];
            pk.h2[1] = (half_t)src[(size_t)(c + 1) * NPTS];
            *(uint32_t*)(dst + c) = pk.u;
        }
    }

    // ---- stage 1: Y = relu(Wsh @ X + biasY), K=128 ----
#pragma unroll
    for (int mt = 0; mt < 4; ++mt) {
        v8f acc = {0.f, 0.f, 0.f, 0.f, 0.f, 0.f, 0.f, 0.f};
#pragma unroll
        for (int kc = 0; kc < 4; ++kc) {
            const char* fb = smem + OFF_WSH + (mt * 4 + kc) * 1024;
            const half_t* xr = Xt + (nbase + ln) * CINP + kc * 32 + hi * 16;
            acc = wmma_f16(load_afrag(fb, ga0, ga1), load_bfrag(xr), acc);
        }
        union { half_t h8[8]; v4u q; } pk;
        const float* bY = biasY + mt * 16 + hi * 8;
#pragma unroll
        for (int r = 0; r < 8; ++r)
            pk.h8[r] = (half_t)fmaxf(acc[r] + bY[r], 0.0f);
        *(v4u*)(Yt + (nbase + ln) * HCP + mt * 16 + hi * 8) = pk.q;
    }

    // ---- stages 2+3 per head ----
    for (int h = 0; h < 6; ++h) {
        // stage 2: Z = relu(W1[h] @ Y + biasZ[h]), K=64
#pragma unroll
        for (int mt = 0; mt < 4; ++mt) {
            v8f acc = {0.f, 0.f, 0.f, 0.f, 0.f, 0.f, 0.f, 0.f};
#pragma unroll
            for (int kc = 0; kc < 2; ++kc) {
                const char* fb = smem + OFF_WH1 + h * 8192 + (mt * 2 + kc) * 1024;
                const half_t* yr = Yt + (nbase + ln) * HCP + kc * 32 + hi * 16;
                acc = wmma_f16(load_afrag(fb, ga0, ga1), load_bfrag(yr), acc);
            }
            union { half_t h8[8]; v4u q; } pk;
            const float* bZ = biasZ + h * HC + mt * 16 + hi * 8;
#pragma unroll
            for (int r = 0; r < 8; ++r)
                pk.h8[r] = (half_t)fmaxf(acc[r] + bZ[r], 0.0f);
            *(v4u*)(Zt + (nbase + ln) * HCP + mt * 16 + hi * 8) = pk.q;
        }

        // stage 3: out = Wf[h] @ Z + bf[h]  (M padded to 16, K=64)
        v8f acc = {0.f, 0.f, 0.f, 0.f, 0.f, 0.f, 0.f, 0.f};
#pragma unroll
        for (int kc = 0; kc < 2; ++kc) {
            const char* fb = smem + OFF_WF + h * 2048 + kc * 1024;
            const half_t* zr = Zt + (nbase + ln) * HCP + kc * 32 + hi * 16;
            acc = wmma_f16(load_afrag(fb, ga0, ga1), load_bfrag(zr), acc);
        }
        const int co = d_COUT[h];
        const int ho = d_HOFF[h];
        float* orow = p.out + ((size_t)b * 12 + ho) * NPTS + n0 + nbase + ln;
#pragma unroll
        for (int r = 0; r < 8; ++r) {
            int m = r + hi * 8;
            if (m < co)
                orow[(size_t)m * NPTS] = acc[r] + biasF[h * 16 + m];
        }
    }
}

extern "C" void kernel_launch(void* const* d_in, const int* in_sizes, int n_in,
                              void* d_out, int out_size, void* d_ws, size_t ws_size,
                              hipStream_t stream) {
    (void)in_sizes; (void)n_in; (void)d_ws; (void)ws_size; (void)out_size;

    Params p;
    p.ct      = (const float*)d_in[0];
    p.sh_w    = (const float*)d_in[1];
    p.sh_b    = (const float*)d_in[2];
    p.sh_g    = (const float*)d_in[3];
    p.sh_beta = (const float*)d_in[4];
    p.sh_mean = (const float*)d_in[5];
    p.sh_var  = (const float*)d_in[6];
    p.h_w1    = (const float*)d_in[7];
    p.h_b1    = (const float*)d_in[8];
    p.h_g1    = (const float*)d_in[9];
    p.h_beta1 = (const float*)d_in[10];
    p.h_mean1 = (const float*)d_in[11];
    p.h_var1  = (const float*)d_in[12];
    for (int h = 0; h < 6; ++h) {
        p.fw[h] = (const float*)d_in[13 + 2 * h];
        p.fb[h] = (const float*)d_in[14 + 2 * h];
    }
    p.out = (float*)d_out;

    hipFuncSetAttribute((const void*)center_head_fused,
                        hipFuncAttributeMaxDynamicSharedMemorySize, SMEM_BYTES);

    dim3 grid(NPTS / TN, B_SZ);
    center_head_fused<<<grid, 256, SMEM_BYTES, stream>>>(p);
}
